// BidirectionalCrossAttention_62740882260663
// MI455X (gfx1250) — compile-verified, matrix operand order checked by reading, not run on hardware
//
#include <hip/hip_runtime.h>

// ---------------------------------------------------------------------------
// Bidirectional cross attention for MI455X (gfx1250), bf16 WMMA pipeline.
// b=2, n=2048, d=1024, heads=16, dim_head=64.
// out      = FA(Q=qk,  K=cqk, V=cv)   (row softmax)
// ctx_out  = FA(Q=cqk, K=qk,  V=v)    (col softmax == row softmax of sim^T)
// ---------------------------------------------------------------------------

typedef __attribute__((ext_vector_type(16))) __bf16 v16bf;
typedef __attribute__((ext_vector_type(8)))  __bf16 v8bf;
typedef __attribute__((ext_vector_type(8)))  float  v8f;

#define NSEQ   2048
#define DMODEL 1024
#define DH     64
#define NROWS  4096   /* b * n */

__device__ __forceinline__ __bf16 f2bf(float f) {
  unsigned u = __builtin_bit_cast(unsigned, f);
  u += 0x7FFFu + ((u >> 16) & 1u);            // round-to-nearest-even
  unsigned short s = (unsigned short)(u >> 16);
  return __builtin_bit_cast(__bf16, s);
}

// Load a 16x32 bf16 A-fragment (or 32x16 B-fragment) for this lane:
// two contiguous 8-element (16B) chunks at K offsets {koff, koff+16}.
__device__ __forceinline__ v16bf load_frag(const __bf16* p0, const __bf16* p1) {
  v16bf r;
  ((v8bf*)&r)[0] = *(const v8bf*)p0;
  ((v8bf*)&r)[1] = *(const v8bf*)p1;
  return r;
}

// Max-reduce across the 16 lanes of each half-wave (N dim of a C tile).
__device__ __forceinline__ void reduce_max16(v8f& v) {
#pragma unroll
  for (int off = 8; off; off >>= 1)
#pragma unroll
    for (int r = 0; r < 8; ++r)
      v[r] = fmaxf(v[r], __shfl_xor(v[r], off));
}

// ---------------------------------------------------------------------------
// Elementwise converts / transposes
// ---------------------------------------------------------------------------
__global__ __launch_bounds__(256)
void k_f32_to_bf16(const float* __restrict__ in, __bf16* __restrict__ out, int n4) {
  int i = blockIdx.x * blockDim.x + threadIdx.x;
  if (i >= n4) return;
  float4 f = ((const float4*)in)[i];
  __bf16* o = out + 4 * (size_t)i;
  o[0] = f2bf(f.x); o[1] = f2bf(f.y); o[2] = f2bf(f.z); o[3] = f2bf(f.w);
}

// WT[n][k] = W[k][n], bf16 output (1024x1024)
__global__ __launch_bounds__(256)
void k_transpose_w(const float* __restrict__ W, __bf16* __restrict__ WT) {
  int idx = blockIdx.x * blockDim.x + threadIdx.x;   // 0 .. 1M-1
  int n = idx >> 10, k = idx & 1023;
  WT[idx] = f2bf(W[(size_t)k * DMODEL + n]);
}

// T[((b*16+h)*64 + d)*2048 + j] = Am[(b*2048+j)*1024 + h*64 + d]
__global__ __launch_bounds__(256)
void k_transpose_act(const __bf16* __restrict__ Am, __bf16* __restrict__ T) {
  int idx = blockIdx.x * blockDim.x + threadIdx.x;   // 0 .. 4M-1
  int j  = idx & (NSEQ - 1);
  int d  = (idx >> 11) & 63;
  int bh = idx >> 17;                 // 0..31
  int b = bh >> 4, h = bh & 15;
  T[idx] = Am[(size_t)(b * NSEQ + j) * DMODEL + h * DH + d];
}

// ---------------------------------------------------------------------------
// GEMM: C(4096x1024) = A(4096x1024, bf16) @ B given pre-transposed
// (BT[n][k], bf16). Each wave computes a 32(M) x 128(N) tile (two M-subtiles
// share every B fragment); 8 waves/WG cover N=1024; grid.x = M/32 = 128.
// ---------------------------------------------------------------------------
template <bool F32OUT>
__global__ __launch_bounds__(256)
void k_gemm(const __bf16* __restrict__ A, const __bf16* __restrict__ BT,
            void* __restrict__ Cout) {
  const int lane = threadIdx.x & 31;
  const int wave = threadIdx.x >> 5;
  const int row0 = blockIdx.x * 32;
  const int n0   = wave * 128;
  const int l15  = lane & 15;
  const int koff = (lane >> 4) * 8;

  const __bf16* arow0 = A + (size_t)(row0 + l15) * DMODEL;
  const __bf16* arow1 = arow0 + (size_t)16 * DMODEL;

  v8f acc[2][8];
#pragma unroll
  for (int s = 0; s < 2; ++s)
#pragma unroll
    for (int t = 0; t < 8; ++t) acc[s][t] = (v8f){};

  for (int k0 = 0; k0 < DMODEL; k0 += 32) {
    __builtin_prefetch(arow0 + k0 + 128, 0, 0);
    v16bf a0 = load_frag(arow0 + k0 + koff, arow0 + k0 + koff + 16);
    v16bf a1 = load_frag(arow1 + k0 + koff, arow1 + k0 + koff + 16);
#pragma unroll
    for (int t = 0; t < 8; ++t) {
      const __bf16* br = BT + (size_t)(n0 + t * 16 + l15) * DMODEL + k0;
      v16bf bf = load_frag(br + koff, br + koff + 16);
      acc[0][t] = __builtin_amdgcn_wmma_f32_16x16x32_bf16(
          false, a0, false, bf, (short)0, acc[0][t], false, false);
      acc[1][t] = __builtin_amdgcn_wmma_f32_16x16x32_bf16(
          false, a1, false, bf, (short)0, acc[1][t], false, false);
    }
  }

  const int rhi = (lane >> 4) << 3;   // +0 or +8 row offset per C layout
#pragma unroll
  for (int s = 0; s < 2; ++s)
#pragma unroll
    for (int t = 0; t < 8; ++t)
#pragma unroll
      for (int r = 0; r < 8; ++r) {
        size_t idx =
            (size_t)(row0 + s * 16 + rhi + r) * DMODEL + n0 + t * 16 + l15;
        if (F32OUT) ((float*)Cout)[idx] = acc[s][t][r];
        else        ((__bf16*)Cout)[idx] = f2bf(acc[s][t][r]);
      }
}

// ---------------------------------------------------------------------------
// Flash attention (online softmax, 64 keys per step). Qm/Km: merged
// (4096x1024) bf16 head projections; VT: (b,h,64,2048) bf16 transposed
// values; Om: merged bf16 out. Grid: (16 i-tiles, 16 heads, 2 batch);
// 8 waves/WG, each wave owns 16 query rows.
// Row sums are computed with a P @ ones WMMA (no shuffle tree); softmax is
// tracked in the exp2 domain so v_exp_f32 is used natively.
// ---------------------------------------------------------------------------
__global__ __launch_bounds__(256)
void k_flash(const __bf16* __restrict__ Qm, const __bf16* __restrict__ Km,
             const __bf16* __restrict__ VT, __bf16* __restrict__ Om,
             float scale) {
  __shared__ alignas(16) __bf16 lds[8][16][64];   // per-wave P staging tile

  const int lane = threadIdx.x & 31;
  const int wave = threadIdx.x >> 5;
  const int it = blockIdx.x, h = blockIdx.y, b = blockIdx.z;
  const int i0  = it * 128 + wave * 16;
  const int l15 = lane & 15;
  const int koff = (lane >> 4) * 8;
  const int rhi  = (lane >> 4) << 3;
  const float c = scale * 1.44269504088896340736f;  // scale * log2(e)

  // Q fragments: 16x64 = two 16x32 A-fragments over dim_head
  const __bf16* qrow = Qm + (size_t)(b * NSEQ + i0 + l15) * DMODEL + h * DH;
  const v16bf q0 = load_frag(qrow + koff,      qrow + koff + 16);
  const v16bf q1 = load_frag(qrow + 32 + koff, qrow + 48 + koff);

  v16bf ones;
#pragma unroll
  for (int i = 0; i < 16; ++i) ones[i] = f2bf(1.0f);

  v8f acc[4];
#pragma unroll
  for (int t = 0; t < 4; ++t) acc[t] = (v8f){};
  v8f m, l;
#pragma unroll
  for (int r = 0; r < 8; ++r) { m[r] = -1e30f; l[r] = 0.f; }

  const __bf16* kbase = Km + (size_t)(b * NSEQ) * DMODEL + h * DH;
  const __bf16* vtb   = VT + (size_t)((b * 16 + h) * DH) * NSEQ;

  for (int j0 = 0; j0 < NSEQ; j0 += 64) {
    // ---- S = Q @ K^T for 64 keys (four 16-key groups), raw scores --------
    v8f s[4];
#pragma unroll
    for (int g = 0; g < 4; ++g) {
      const __bf16* krow = kbase + (size_t)(j0 + g * 16 + l15) * DMODEL;
      v16bf kb0 = load_frag(krow + koff,      krow + koff + 16);
      v16bf kb1 = load_frag(krow + 32 + koff, krow + 48 + koff);
      v8f z = (v8f){};
      z = __builtin_amdgcn_wmma_f32_16x16x32_bf16(false, q0, false, kb0,
                                                  (short)0, z, false, false);
      s[g] = __builtin_amdgcn_wmma_f32_16x16x32_bf16(false, q1, false, kb1,
                                                     (short)0, z, false, false);
    }

    // ---- running max (raw-score domain), one lane reduction per 64 keys --
    v8f mx;
#pragma unroll
    for (int r = 0; r < 8; ++r)
      mx[r] = fmaxf(fmaxf(s[0][r], s[1][r]), fmaxf(s[2][r], s[3][r]));
    reduce_max16(mx);

    v8f alpha, mc;
#pragma unroll
    for (int r = 0; r < 8; ++r) {
      float mn = fmaxf(m[r], mx[r]);
      alpha[r] = exp2f((m[r] - mn) * c);
      m[r] = mn;
      mc[r] = mn * c;
    }
    // p = 2^(s*c - m*c), staged to LDS as bf16 in A-layout
#pragma unroll
    for (int g = 0; g < 4; ++g)
#pragma unroll
      for (int r = 0; r < 8; ++r)
        lds[wave][rhi + r][l15 + g * 16] =
            f2bf(exp2f(fmaf(s[g][r], c, -mc[r])));

#pragma unroll
    for (int t = 0; t < 4; ++t)
#pragma unroll
      for (int r = 0; r < 8; ++r) acc[t][r] *= alpha[r];

    asm volatile("s_wait_dscnt 0" ::: "memory");
    v16bf pA0 = load_frag(&lds[wave][l15][koff],      &lds[wave][l15][koff + 16]);
    v16bf pA1 = load_frag(&lds[wave][l15][32 + koff], &lds[wave][l15][48 + koff]);

    // ---- row sums of the bf16 P actually used: lsum = P @ ones -----------
    v8f z = (v8f){};
    z = __builtin_amdgcn_wmma_f32_16x16x32_bf16(false, pA0, false, ones,
                                                (short)0, z, false, false);
    z = __builtin_amdgcn_wmma_f32_16x16x32_bf16(false, pA1, false, ones,
                                                (short)0, z, false, false);
#pragma unroll
    for (int r = 0; r < 8; ++r) l[r] = l[r] * alpha[r] + z[r];

    // ---- acc += P(16x64) @ V(64x64), V pre-transposed (d-major) ----------
#pragma unroll
    for (int t = 0; t < 4; ++t) {
      const __bf16* vr = vtb + (size_t)(t * 16 + l15) * NSEQ + j0;
      v16bf vb0 = load_frag(vr + koff,      vr + koff + 16);
      v16bf vb1 = load_frag(vr + 32 + koff, vr + 48 + koff);
      acc[t] = __builtin_amdgcn_wmma_f32_16x16x32_bf16(
          false, pA0, false, vb0, (short)0, acc[t], false, false);
      acc[t] = __builtin_amdgcn_wmma_f32_16x16x32_bf16(
          false, pA1, false, vb1, (short)0, acc[t], false, false);
    }
  }

  // ---- normalize and store into merged (4096x1024) layout ----------------
  v8f rl;
#pragma unroll
  for (int r = 0; r < 8; ++r) rl[r] = 1.0f / l[r];
  __bf16* obase = Om + (size_t)(b * NSEQ) * DMODEL + h * DH;
#pragma unroll
  for (int t = 0; t < 4; ++t)
#pragma unroll
    for (int r = 0; r < 8; ++r)
      obase[(size_t)(i0 + rhi + r) * DMODEL + t * 16 + l15] =
          f2bf(acc[t][r] * rl[r]);
}

// ---------------------------------------------------------------------------
// Host launcher
// ---------------------------------------------------------------------------
extern "C" void kernel_launch(void* const* d_in, const int* in_sizes, int n_in,
                              void* d_out, int out_size, void* d_ws, size_t ws_size,
                              hipStream_t stream) {
  (void)in_sizes; (void)n_in; (void)out_size; (void)ws_size;
  const float* x      = (const float*)d_in[0];
  const float* ctx    = (const float*)d_in[1];
  const float* w_qk   = (const float*)d_in[2];
  const float* w_v    = (const float*)d_in[3];
  const float* w_cqk  = (const float*)d_in[4];
  const float* w_cv   = (const float*)d_in[5];
  const float* w_out  = (const float*)d_in[6];
  const float* w_cout = (const float*)d_in[7];

  const size_t ACT = (size_t)NROWS * DMODEL;   // 4M elems
  const size_t WEL = (size_t)DMODEL * DMODEL;  // 1M elems

  __bf16* xb    = (__bf16*)d_ws;
  __bf16* cb    = xb + ACT;
  __bf16* qk    = xb + 2 * ACT;
  __bf16* vv    = xb + 3 * ACT;
  __bf16* cqk   = xb + 4 * ACT;
  __bf16* cv    = xb + 5 * ACT;
  __bf16* vT    = xb + 6 * ACT;
  __bf16* cvT   = xb + 7 * ACT;
  __bf16* om1   = xb + 8 * ACT;
  __bf16* om2   = xb + 9 * ACT;
  __bf16* wqkT  = xb + 10 * ACT;
  __bf16* wvT   = wqkT + WEL;
  __bf16* wcqkT = wqkT + 2 * WEL;
  __bf16* wcvT  = wqkT + 3 * WEL;
  __bf16* woutT = wqkT + 4 * WEL;
  __bf16* wcoutT= wqkT + 5 * WEL;

  const int B = 256;
  // 1) converts
  k_f32_to_bf16<<<(int)(ACT / 4 / B), B, 0, stream>>>(x,   xb, (int)(ACT / 4));
  k_f32_to_bf16<<<(int)(ACT / 4 / B), B, 0, stream>>>(ctx, cb, (int)(ACT / 4));
  k_transpose_w<<<(int)(WEL / B), B, 0, stream>>>(w_qk,   wqkT);
  k_transpose_w<<<(int)(WEL / B), B, 0, stream>>>(w_v,    wvT);
  k_transpose_w<<<(int)(WEL / B), B, 0, stream>>>(w_cqk,  wcqkT);
  k_transpose_w<<<(int)(WEL / B), B, 0, stream>>>(w_cv,   wcvT);
  k_transpose_w<<<(int)(WEL / B), B, 0, stream>>>(w_out,  woutT);
  k_transpose_w<<<(int)(WEL / B), B, 0, stream>>>(w_cout, wcoutT);
  // 2) projections (bf16 out)
  k_gemm<false><<<NROWS / 32, B, 0, stream>>>(xb, wqkT,  qk);
  k_gemm<false><<<NROWS / 32, B, 0, stream>>>(xb, wvT,   vv);
  k_gemm<false><<<NROWS / 32, B, 0, stream>>>(cb, wcqkT, cqk);
  k_gemm<false><<<NROWS / 32, B, 0, stream>>>(cb, wcvT,  cv);
  // 3) value transposes to (b,h,64,2048)
  k_transpose_act<<<(int)(ACT / B), B, 0, stream>>>(vv, vT);
  k_transpose_act<<<(int)(ACT / B), B, 0, stream>>>(cv, cvT);
  // 4) both attention directions
  const float scale = 0.125f;   // 64^-0.5
  dim3 fg(NSEQ / 128, 16, 2);
  k_flash<<<fg, B, 0, stream>>>(qk,  cqk, cvT, om1, scale);
  k_flash<<<fg, B, 0, stream>>>(cqk, qk,  vT,  om2, scale);
  // 5) output projections -> fp32 d_out (out, then context_out)
  float* outp = (float*)d_out;
  k_gemm<true><<<NROWS / 32, B, 0, stream>>>(om1, woutT,  (void*)outp);
  k_gemm<true><<<NROWS / 32, B, 0, stream>>>(om2, wcoutT, (void*)(outp + ACT));
}
